// GNN_node_33457795236710
// MI455X (gfx1250) — matrix operand order repaired and checked by reading, not compiled, not run on hardware
//
#include <hip/hip_runtime.h>

#define NN 100000
#define EE 500000
#define DD 128
#define LL 5
#define ATOM_F 9
#define ATOM_V 120
#define BOND_F 3
#define BOND_V 8

typedef __attribute__((ext_vector_type(16))) __bf16        v16bf;
typedef __attribute__((ext_vector_type(2)))  __bf16        v2bf;
typedef __attribute__((ext_vector_type(8)))  float         v8f;
typedef __attribute__((ext_vector_type(8)))  unsigned int  v8u;
typedef __attribute__((ext_vector_type(4)))  unsigned int  v4u;

union ABfrag { v8u u8; v4u u4[2]; v16bf b; };

__device__ __forceinline__ unsigned int pack2(float lo, float hi) {
    union { v2bf v; unsigned int u; } p;
    p.v[0] = (__bf16)lo;
    p.v[1] = (__bf16)hi;
    return p.u;
}

// ---------------- h init: h[n][c] = sum_f atom_emb[f][x[n][f]][c] ----------------
__global__ void init_h_kernel(const int* __restrict__ x,
                              const float* __restrict__ atom_emb,
                              float* __restrict__ h) {
    const int node = blockIdx.x;
    const int c = threadIdx.x;            // 128 threads
    float s = 0.0f;
    #pragma unroll
    for (int f = 0; f < ATOM_F; ++f) {
        const int idx = x[node * ATOM_F + f];        // wave-uniform
        s += atom_emb[((size_t)(f * ATOM_V + idx)) * DD + c];
    }
    h[(size_t)node * DD + c] = s;
}

// ---------------- zero ----------------
__global__ void zero_kernel(float* __restrict__ p, size_t n) {
    size_t i = (size_t)blockIdx.x * blockDim.x + threadIdx.x;
    if (i < n) p[i] = 0.0f;
}

// ---- edge phase: one wave per edge, float4 per lane ----
__global__ void edge_kernel(const int* __restrict__ edge_index,
                            const int* __restrict__ edge_attr,
                            const float* __restrict__ edge_weight,
                            const float* __restrict__ bond_emb_l,   // [3][8][128]
                            const float* __restrict__ h,
                            float* __restrict__ agg) {
    const size_t t = (size_t)blockIdx.x * blockDim.x + threadIdx.x;
    const int e  = (int)(t >> 5);
    const int c4 = (int)(t & 31) * 4;
    if (e >= EE) return;
    const int src = edge_index[e];          // wave-uniform
    const int dst = edge_index[EE + e];
    const float ew = edge_weight[e];
    const int a0 = edge_attr[e * 3 + 0];
    const int a1 = edge_attr[e * 3 + 1];
    const int a2 = edge_attr[e * 3 + 2];
    const float4 e0 = *(const float4*)(bond_emb_l + (size_t)(0 * BOND_V + a0) * DD + c4);
    const float4 e1 = *(const float4*)(bond_emb_l + (size_t)(1 * BOND_V + a1) * DD + c4);
    const float4 e2 = *(const float4*)(bond_emb_l + (size_t)(2 * BOND_V + a2) * DD + c4);
    const float4 hv = *(const float4*)(h + (size_t)src * DD + c4);
    float m0 = hv.x + e0.x + e1.x + e2.x;
    float m1 = hv.y + e0.y + e1.y + e2.y;
    float m2 = hv.z + e0.z + e1.z + e2.z;
    float m3 = hv.w + e0.w + e1.w + e2.w;
    m0 = (m0 > 0.0f ? m0 : 0.0f) * ew;
    m1 = (m1 > 0.0f ? m1 : 0.0f) * ew;
    m2 = (m2 > 0.0f ? m2 : 0.0f) * ew;
    m3 = (m3 > 0.0f ? m3 : 0.0f) * ew;
    float* ap = agg + (size_t)dst * DD + c4;
    atomicAdd(ap + 0, m0);
    atomicAdd(ap + 1, m1);
    atomicAdd(ap + 2, m2);
    atomicAdd(ap + 3, m3);
}

// ---- fragment-major weight pre-pack ----
// Layout: [l][kk][ntile][lane][v]  (v8u per lane, contiguous 32B)
//   n = ntile*16 + (lane&15) ; k = kk*32 + (lane>>4)*16 + 2v ; pack(W[k][n], W[k+1][n])
__global__ void pack_w_frag_kernel(const float* __restrict__ W, unsigned int* __restrict__ Wp,
                                   int K, int Ncol, int layers) {
    const int KK = K / 32, NT = Ncol / 16;
    const int per = KK * NT * 256;
    const int i = blockIdx.x * blockDim.x + threadIdx.x;
    if (i >= per * layers) return;
    const int l  = i / per;
    int r        = i - l * per;
    const int kk = r / (NT * 256);  r -= kk * NT * 256;
    const int nt = r / 256;         r -= nt * 256;
    const int lane = r >> 3;
    const int v    = r & 7;
    const int n = nt * 16 + (lane & 15);
    const int k = kk * 32 + (lane >> 4) * 16 + 2 * v;
    const float* Wl = W + (size_t)l * K * Ncol;
    Wp[i] = pack2(Wl[(size_t)k * Ncol + n], Wl[(size_t)(k + 1) * Ncol + n]);
}

// ---------------- GEMM1: t1 = ((1+eps)*h + agg) @ W1 + b1 ; fused column stats ----------------
__launch_bounds__(256, 2)
__global__ void gemm1_kernel(const float* __restrict__ h,
                             const float* __restrict__ agg,
                             const float* __restrict__ eps, int layer,
                             const unsigned int* __restrict__ Bp,   // frag-major [4][16][32][8]
                             const float* __restrict__ bias,        // [256]
                             float* __restrict__ t1,                // [N][256]
                             float* __restrict__ stat_sum,
                             float* __restrict__ stat_sq) {
    __shared__ unsigned int Asp[64][68];  // packed bf16 pairs, 16B-aligned padded stride
    __shared__ float ssum[256];
    __shared__ float ssq[256];

    const int tid   = threadIdx.x;
    const int lane  = tid & 31;
    const int wave  = tid >> 5;
    const int rA    = lane & 15;
    const int khalf = lane >> 4;
    const int mw    = wave >> 1;      // 0..3 : row group
    const int nw    = wave & 1;       // 0..1 : col half
    const int m0    = blockIdx.x * 64;
    const float e1  = 1.0f + eps[layer];

    ssum[tid] = 0.0f;
    ssq[tid]  = 0.0f;

    { // stage A (z = (1+eps)*h + agg): 64 rows x 64 pairs, float4 global loads
        const int row = tid >> 2;
        const int k2b = (tid & 3) * 16;
        const int gr  = m0 + row;
        if (gr < NN) {
            const float4* hp = (const float4*)(h   + (size_t)gr * DD);
            const float4* ap = (const float4*)(agg + (size_t)gr * DD);
            #pragma unroll
            for (int i = 0; i < 8; ++i) {
                float4 hv = hp[(k2b >> 1) + i];
                float4 av = ap[(k2b >> 1) + i];
                Asp[row][k2b + 2 * i]     = pack2(e1 * hv.x + av.x, e1 * hv.y + av.y);
                Asp[row][k2b + 2 * i + 1] = pack2(e1 * hv.z + av.z, e1 * hv.w + av.w);
            }
        } else {
            #pragma unroll
            for (int i = 0; i < 16; ++i) Asp[row][k2b + i] = 0u;
        }
    }
    __syncthreads();

    const v8f vz = {0.f, 0.f, 0.f, 0.f, 0.f, 0.f, 0.f, 0.f};
    v8f acc[8];
    #pragma unroll
    for (int i = 0; i < 8; ++i) acc[i] = vz;

    #pragma unroll
    for (int kk = 0; kk < 4; ++kk) {          // K = 128 = 4 x 32
        const int K0h = kk * 16;
        ABfrag a;
        a.u4[0] = *(const v4u*)&Asp[mw * 16 + rA][K0h + 8 * khalf];
        a.u4[1] = *(const v4u*)&Asp[mw * 16 + rA][K0h + 8 * khalf + 4];
        #pragma unroll
        for (int nt = 0; nt < 8; ++nt) {
            const int ntile = nw * 8 + nt;               // NT = 16
            ABfrag bf;
            bf.u8 = *(const v8u*)(Bp + ((size_t)((kk * 16 + ntile) * 32 + lane) << 3));
            acc[nt] = __builtin_amdgcn_wmma_f32_16x16x32_bf16(
                false, a.b, false, bf.b, (short)0, acc[nt], false, false);
        }
    }

    #pragma unroll
    for (int nt = 0; nt < 8; ++nt) {
        const int col = nw * 128 + nt * 16 + rA;
        const float bb = bias[col];
        float s1 = 0.0f, s2 = 0.0f;
        #pragma unroll
        for (int r = 0; r < 8; ++r) {
            const int grow = m0 + mw * 16 + khalf * 8 + r;
            if (grow < NN) {
                float v = acc[nt][r] + bb;
                t1[(size_t)grow * 256 + col] = v;
                s1 += v;
                s2 += v * v;
            }
        }
        atomicAdd(&ssum[col], s1);
        atomicAdd(&ssq[col], s2);
    }
    __syncthreads();
    atomicAdd(&stat_sum[tid], ssum[tid]);
    atomicAdd(&stat_sq[tid],  ssq[tid]);
}

// ---------------- finalize BN: fold (gamma,beta,mu,sigma) into scale/shift ----------------
__global__ void finalize_bn_kernel(const float* __restrict__ sum,
                                   const float* __restrict__ sq,
                                   const float* __restrict__ g,
                                   const float* __restrict__ bt,
                                   float* __restrict__ scale,
                                   float* __restrict__ shift, int C) {
    const int t = threadIdx.x;
    if (t >= C) return;
    const float inv = 1.0f / (float)NN;
    const float mu  = sum[t] * inv;
    const float var = sq[t] * inv - mu * mu;
    const float rs  = rsqrtf(var + 1e-5f);
    const float a   = g[t] * rs;
    scale[t] = a;
    shift[t] = bt[t] - mu * a;
}

// ---------------- GEMM2: t2 = relu(bn1(t1)) @ W2 + b2 ; fused column stats ----------------
__launch_bounds__(256, 2)
__global__ void gemm2_kernel(const float* __restrict__ t1,          // [N][256]
                             const float* __restrict__ scale1,      // [256]
                             const float* __restrict__ shift1,      // [256]
                             const unsigned int* __restrict__ Bp,   // frag-major [8][8][32][8]
                             const float* __restrict__ bias,        // [128]
                             float* __restrict__ t2,                // [N][128]
                             float* __restrict__ stat_sum,
                             float* __restrict__ stat_sq) {
    __shared__ unsigned int Asp[64][132]; // 16B-aligned padded stride
    __shared__ float sc[256];
    __shared__ float sh[256];
    __shared__ float ssum[128];
    __shared__ float ssq[128];

    const int tid   = threadIdx.x;
    const int lane  = tid & 31;
    const int wave  = tid >> 5;
    const int rA    = lane & 15;
    const int khalf = lane >> 4;
    const int mw    = wave >> 1;
    const int nw    = wave & 1;
    const int m0    = blockIdx.x * 64;

    sc[tid] = scale1[tid];
    sh[tid] = shift1[tid];
    if (tid < 128) { ssum[tid] = 0.0f; ssq[tid] = 0.0f; }
    __syncthreads();

    { // stage A = relu(bn1(t1)) : 64 rows x 128 pairs, float4 global loads
        const int row = tid >> 2;
        const int k2b = (tid & 3) * 32;
        const int gr  = m0 + row;
        if (gr < NN) {
            const float4* tp = (const float4*)(t1 + (size_t)gr * 256);
            #pragma unroll
            for (int i = 0; i < 16; ++i) {
                const int c = 2 * k2b + 4 * i;
                float4 tv = tp[(k2b >> 1) + i];
                float v0 = sc[c]     * tv.x + sh[c];     v0 = v0 > 0.0f ? v0 : 0.0f;
                float v1 = sc[c + 1] * tv.y + sh[c + 1]; v1 = v1 > 0.0f ? v1 : 0.0f;
                float v2 = sc[c + 2] * tv.z + sh[c + 2]; v2 = v2 > 0.0f ? v2 : 0.0f;
                float v3 = sc[c + 3] * tv.w + sh[c + 3]; v3 = v3 > 0.0f ? v3 : 0.0f;
                Asp[row][k2b + 2 * i]     = pack2(v0, v1);
                Asp[row][k2b + 2 * i + 1] = pack2(v2, v3);
            }
        } else {
            #pragma unroll
            for (int i = 0; i < 32; ++i) Asp[row][k2b + i] = 0u;
        }
    }
    __syncthreads();

    const v8f vz = {0.f, 0.f, 0.f, 0.f, 0.f, 0.f, 0.f, 0.f};
    v8f acc[4];
    #pragma unroll
    for (int i = 0; i < 4; ++i) acc[i] = vz;

    #pragma unroll
    for (int kk = 0; kk < 8; ++kk) {          // K = 256 = 8 x 32
        const int K0h = kk * 16;
        ABfrag a;
        a.u4[0] = *(const v4u*)&Asp[mw * 16 + rA][K0h + 8 * khalf];
        a.u4[1] = *(const v4u*)&Asp[mw * 16 + rA][K0h + 8 * khalf + 4];
        #pragma unroll
        for (int nt = 0; nt < 4; ++nt) {
            const int ntile = nw * 4 + nt;               // NT = 8
            ABfrag bf;
            bf.u8 = *(const v8u*)(Bp + ((size_t)((kk * 8 + ntile) * 32 + lane) << 3));
            acc[nt] = __builtin_amdgcn_wmma_f32_16x16x32_bf16(
                false, a.b, false, bf.b, (short)0, acc[nt], false, false);
        }
    }

    #pragma unroll
    for (int nt = 0; nt < 4; ++nt) {
        const int col = nw * 64 + nt * 16 + rA;
        const float bb = bias[col];
        float s1 = 0.0f, s2 = 0.0f;
        #pragma unroll
        for (int r = 0; r < 8; ++r) {
            const int grow = m0 + mw * 16 + khalf * 8 + r;
            if (grow < NN) {
                float v = acc[nt][r] + bb;
                t2[(size_t)grow * 128 + col] = v;
                s1 += v;
                s2 += v * v;
            }
        }
        atomicAdd(&ssum[col], s1);
        atomicAdd(&ssq[col], s2);
    }
    __syncthreads();
    if (tid < 128) {
        atomicAdd(&stat_sum[tid], ssum[tid]);
        atomicAdd(&stat_sq[tid],  ssq[tid]);
    }
}

// ---------------- output BN (+optional relu) ----------------
__global__ void bn_out_kernel(const float* __restrict__ t2,
                              const float* __restrict__ scale,
                              const float* __restrict__ shift,
                              float* __restrict__ out, int do_relu) {
    const size_t i = (size_t)blockIdx.x * blockDim.x + threadIdx.x;
    if (i >= (size_t)NN * DD) return;
    const int c = (int)(i & (DD - 1));
    float v = scale[c] * t2[i] + shift[c];
    if (do_relu) v = v > 0.0f ? v : 0.0f;
    out[i] = v;
}

extern "C" void kernel_launch(void* const* d_in, const int* in_sizes, int n_in,
                              void* d_out, int out_size, void* d_ws, size_t ws_size,
                              hipStream_t stream) {
    (void)in_sizes; (void)n_in; (void)out_size; (void)ws_size;
    const int*   x           = (const int*)d_in[0];
    const int*   edge_index  = (const int*)d_in[1];
    const int*   edge_attr   = (const int*)d_in[2];
    const float* edge_weight = (const float*)d_in[3];
    const float* atom_emb    = (const float*)d_in[4];
    const float* bond_emb    = (const float*)d_in[5];
    const float* eps         = (const float*)d_in[6];
    const float* W1          = (const float*)d_in[7];
    const float* b1          = (const float*)d_in[8];
    const float* g1          = (const float*)d_in[9];
    const float* bt1         = (const float*)d_in[10];
    const float* W2          = (const float*)d_in[11];
    const float* b2          = (const float*)d_in[12];
    const float* g_out       = (const float*)d_in[13];
    const float* bt_out      = (const float*)d_in[14];

    const size_t ND = (size_t)NN * DD;
    float* ws     = (float*)d_ws;
    float* hbuf   = ws;                 // N*D
    float* aggbuf = ws + ND;            // N*D  (reused as t2)
    float* t1buf  = ws + 2 * ND;        // N*2D
    float* st     = ws + 4 * ND;        // 2048 floats of stats
    float* sum1 = st,        *sq1 = st + 256,  *scale1 = st + 512,  *shift1 = st + 768;
    float* sum2 = st + 1024, *sq2 = st + 1152, *scale2 = st + 1280, *shift2 = st + 1408;
    unsigned int* W1p = (unsigned int*)(ws + 4 * ND + 2048);   // 5*64*256
    unsigned int* W2p = W1p + 5 * 64 * 256;                    // 5*128*128

    // pre-pack weights to bf16 fragment-major layout (deterministic, every call)
    pack_w_frag_kernel<<<(5 * 64 * 256 + 255) / 256, 256, 0, stream>>>(W1, W1p, 128, 256, 5);
    pack_w_frag_kernel<<<(5 * 128 * 128 + 255) / 256, 256, 0, stream>>>(W2, W2p, 256, 128, 5);

    init_h_kernel<<<NN, 128, 0, stream>>>(x, atom_emb, hbuf);

    const int gemmBlocks = (NN + 63) / 64;
    const int edgeBlocks = (int)(((size_t)EE * 32 + 255) / 256);
    const int ndBlocks   = (int)((ND + 255) / 256);

    for (int l = 0; l < LL; ++l) {
        zero_kernel<<<ndBlocks, 256, 0, stream>>>(aggbuf, ND);
        zero_kernel<<<8, 256, 0, stream>>>(st, 2048);

        edge_kernel<<<edgeBlocks, 256, 0, stream>>>(
            edge_index, edge_attr, edge_weight,
            bond_emb + (size_t)l * BOND_F * BOND_V * DD, hbuf, aggbuf);

        gemm1_kernel<<<gemmBlocks, 256, 0, stream>>>(
            hbuf, aggbuf, eps, l,
            W1p + (size_t)l * 64 * 256, b1 + (size_t)l * 256,
            t1buf, sum1, sq1);

        finalize_bn_kernel<<<1, 256, 0, stream>>>(
            sum1, sq1, g1 + (size_t)l * 256, bt1 + (size_t)l * 256, scale1, shift1, 256);

        gemm2_kernel<<<gemmBlocks, 256, 0, stream>>>(
            t1buf, scale1, shift1,
            W2p + (size_t)l * 128 * 128, b2 + (size_t)l * 128,
            aggbuf /* t2 */, sum2, sq2);

        finalize_bn_kernel<<<1, 128, 0, stream>>>(
            sum2, sq2, g_out + (size_t)l * 128, bt_out + (size_t)l * 128, scale2, shift2, 128);

        float* dst = (l == LL - 1) ? (float*)d_out : hbuf;
        bn_out_kernel<<<ndBlocks, 256, 0, stream>>>(aggbuf, scale2, shift2, dst, l != LL - 1);
    }
}
